// AtnConv_27487790694542
// MI455X (gfx1250) — matrix operand order repaired and checked by reading, not compiled, not run on hardware
//
#include <hip/hip_runtime.h>
#include <cstdint>

// ---------------------------------------------------------------------------
// CDNA5 (gfx1250) contextual-attention pipeline, bf16 WMMA + fp32 accumulate,
// TDM (tensor_load_to_lds) double-buffered staging for the score GEMM.
// ---------------------------------------------------------------------------

typedef unsigned short u16;
typedef __attribute__((ext_vector_type(16))) __bf16 v16bf;
typedef __attribute__((ext_vector_type(8)))  float  v8f;
typedef __attribute__((ext_vector_type(4)))  unsigned int v4u;
typedef __attribute__((ext_vector_type(8)))  int v8i_t;
typedef __attribute__((ext_vector_type(4)))  int v4i_t;

union Frag16 { v16bf v; uint4 q[2]; };
union Acc8   { v8f v; float f[8]; };

#if __has_builtin(__builtin_amdgcn_tensor_load_to_lds) && \
    __has_builtin(__builtin_amdgcn_s_wait_tensorcnt)
#define USE_TDM 1
#else
#define USE_TDM 0
#endif

__device__ inline u16 f2bf(float f) {
  union { float f; unsigned u; } a; a.f = f;
  unsigned r = a.u + 0x7fffu + ((a.u >> 16) & 1u);   // round-to-nearest-even
  return (u16)(r >> 16);
}

#if USE_TDM
// DMA one 2D tile (32 bf16 columns x 128 rows, row stride 1152 elems) -> LDS.
__device__ __forceinline__ void tdm_load_tile(const u16* gsrc, unsigned ldsByteOff)
{
  unsigned long long ga = (unsigned long long)(uintptr_t)gsrc;
  v4u g0;
  g0.x = 1u;                                        // count=1, user descriptor
  g0.y = ldsByteOff;                                // lds_addr (bytes)
  g0.z = (unsigned)ga;                              // global_addr[31:0]
  g0.w = (unsigned)((ga >> 32) & 0x01ffffffu)       // global_addr[56:32]
       | (2u << 30);                                // type=2 ("image")
  v8i_t g1;
  g1[0] = 0x10000;                                  // data_size=1 -> 2 bytes
  g1[1] = (int)(32u  << 16);                        // tensor_dim0 = 32
  g1[2] = (int)(128u << 16);                        // tensor_dim1 = 128
  g1[3] = (int)(32u  << 16);                        // tile_dim0 = 32
  g1[4] = 128;                                      // tile_dim1 = 128, tile_dim2 = 0
  g1[5] = 1152;                                     // tensor_dim0_stride = 1152
  g1[6] = 0;
  g1[7] = 0;
  v4i_t gz = {0, 0, 0, 0};
#if __clang_major__ >= 23
  v8i_t gz8 = {0, 0, 0, 0, 0, 0, 0, 0};
  __builtin_amdgcn_tensor_load_to_lds(g0, g1, gz, gz, gz8, 0);
#else
  __builtin_amdgcn_tensor_load_to_lds(g0, g1, gz, gz, 0);
#endif
}
#endif

// ---------------------------------------------------------------------------
// Kernel 1: im2col of x2 -> bf16 Bmat[s][1152], patch inv-norms, mask flags.
// ---------------------------------------------------------------------------
__global__ void __launch_bounds__(128)
prep_kernel(const float* __restrict__ x2,    // [128][64][64] (one sample)
            const float* __restrict__ mask,  // [64][64]
            u16* __restrict__ Bmat,          // [4096][1152]
            float* __restrict__ invn,        // [4096]
            float* __restrict__ mmf)         // [4096]
{
  __shared__ float red[128];
  int s = blockIdx.x;
  int sy = s >> 6, sx = s & 63;
  int tid = threadIdx.x;
  float ss = 0.f;
  for (int k = tid; k < 1152; k += 128) {
    int c = k / 9, t = k - c * 9;
    int kh = t / 3, kw = t - kh * 3;
    int yy = sy + kh - 1, xx = sx + kw - 1;
    float v = 0.f;
    if ((unsigned)yy < 64u && (unsigned)xx < 64u)
      v = x2[(c << 12) + (yy << 6) + xx];
    Bmat[(size_t)s * 1152 + k] = f2bf(v);
    ss += v * v;
  }
  red[tid] = ss;
  __syncthreads();
  for (int off = 64; off > 0; off >>= 1) {
    if (tid < off) red[tid] += red[tid + off];
    __syncthreads();
  }
  if (tid == 0) {
    invn[s] = 1.0f / fmaxf(sqrtf(red[0]), 1e-4f);
    float msum = 0.f;
    for (int kh = 0; kh < 3; ++kh)
      for (int kw = 0; kw < 3; ++kw) {
        int yy = sy + kh - 1, xx = sx + kw - 1;
        if ((unsigned)yy < 64u && (unsigned)xx < 64u) msum += mask[(yy << 6) + xx];
      }
    mmf[s] = (msum == 0.f) ? 1.f : 0.f;
  }
}

// ---------------------------------------------------------------------------
// Kernel 2: Gram GEMM  Sx[s][p] = sum_k Bmat[s][k]*Bmat[p][k]  (M=N=4096,K=1152)
// 128x128 tile/block; K staged 32-wide via TDM, manually 2x-unrolled double
// buffer (compile-time buffer parity -> no accumulator shuffling).
// ---------------------------------------------------------------------------
__global__ void __launch_bounds__(256)
gemm_scores_kernel(const u16* __restrict__ Bmat, float* __restrict__ Sx)
{
  // lT[0..1] = A double buffer, lT[2..3] = B double buffer (8 KB each).
  __shared__ u16 lT[4][128][32];
  int tid  = threadIdx.x;
  int lane = tid & 31;
  int wv   = tid >> 5;                 // 8 waves
  int s0 = blockIdx.y * 128;
  int p0 = blockIdx.x * 128;
  int srow = (wv & 3) * 32;            // wave covers 32 s-rows (2 M-tiles)
  int ncol = (wv >> 2) * 64;           // and 64 p-cols (4 N-tiles)

  v8f vzero = {0.f,0.f,0.f,0.f,0.f,0.f,0.f,0.f};
  v8f acc[2][4];
#pragma unroll
  for (int i = 0; i < 2; ++i)
#pragma unroll
    for (int j = 0; j < 4; ++j) acc[i][j] = vzero;

  int m   = lane & 15;
  int alb = (lane < 16) ? 0 : 16;      // A-operand lane K-base (contiguous k = 16h + s)
  int bkb = (lane < 16) ? 0 : 16;      // B-operand lane K-base

  const u16* rowA = Bmat + (size_t)s0 * 1152;
  const u16* rowB = Bmat + (size_t)p0 * 1152;

  // 8 WMMAs from LDS buffer parity `cur` (constant-folded when cur literal).
  auto compute = [&](int cur) {
    Frag16 afr[2], bfr[4];
#pragma unroll
    for (int i = 0; i < 2; ++i) {
      const u16* pr = &lT[cur][srow + i * 16 + m][0];
      afr[i].q[0] = *(const uint4*)(pr + alb);        // K lb..lb+7
      afr[i].q[1] = *(const uint4*)(pr + alb + 8);    // K lb+8..lb+15
    }
#pragma unroll
    for (int j = 0; j < 4; ++j) {
      const u16* pr = &lT[2 + cur][ncol + j * 16 + m][0];
      const uint4* pb = (const uint4*)(pr + bkb);     // K kb..kb+15 contiguous
      bfr[j].q[0] = pb[0]; bfr[j].q[1] = pb[1];
    }
#pragma unroll
    for (int i = 0; i < 2; ++i)
#pragma unroll
      for (int j = 0; j < 4; ++j)
        acc[i][j] = __builtin_amdgcn_wmma_f32_16x16x32_bf16(
            false, afr[i].v, false, bfr[j].v, (short)0, acc[i][j], false, false);
  };

  const int NSTEP = 36;                // 1152 / 32 (even)

#if USE_TDM
  unsigned ldsBase = (unsigned)(uintptr_t)&lT[0][0][0];
  if (wv == 0) {
    tdm_load_tile(rowA, ldsBase);
    tdm_load_tile(rowB, ldsBase + 2u * 8192u);
  }
  for (int step = 0; step < NSTEP; step += 2) {
    // --- phase 0: compute buffer 0, prefetch step+1 into buffer 1 ---
    if (wv == 0) {
      tdm_load_tile(rowA + (step + 1) * 32, ldsBase + 1u * 8192u);
      tdm_load_tile(rowB + (step + 1) * 32, ldsBase + 3u * 8192u);
      __builtin_amdgcn_s_wait_tensorcnt((short)2);    // step's pair complete
    }
    __syncthreads();
    compute(0);
    __syncthreads();
    // --- phase 1: compute buffer 1, prefetch step+2 into buffer 0 ---
    if (wv == 0) {
      if (step + 2 < NSTEP) {
        tdm_load_tile(rowA + (step + 2) * 32, ldsBase);
        tdm_load_tile(rowB + (step + 2) * 32, ldsBase + 2u * 8192u);
        __builtin_amdgcn_s_wait_tensorcnt((short)2);  // (step+1)'s pair complete
      } else {
        __builtin_amdgcn_s_wait_tensorcnt((short)0);
      }
    }
    __syncthreads();
    compute(1);
    __syncthreads();
  }
#else
  int lrow = tid >> 1;                 // 128 rows, 2 threads/row
  int lseg = (tid & 1) * 16;           // 16 halfs (32B) each
  for (int step = 0; step < NSTEP; ++step) {
    int k0 = step * 32;
    __syncthreads();
    {
      const uint4* ga = (const uint4*)(rowA + (size_t)lrow * 1152 + k0 + lseg);
      uint4* da = (uint4*)&lT[0][lrow][lseg];
      da[0] = ga[0]; da[1] = ga[1];
      const uint4* gb = (const uint4*)(rowB + (size_t)lrow * 1152 + k0 + lseg);
      uint4* db = (uint4*)&lT[2][lrow][lseg];
      db[0] = gb[0]; db[1] = gb[1];
    }
    __syncthreads();
    compute(0);
  }
#endif

  int n  = lane & 15;
  int mh = (lane < 16) ? 0 : 8;
#pragma unroll
  for (int i = 0; i < 2; ++i)
#pragma unroll
    for (int j = 0; j < 4; ++j) {
      Acc8 a; a.v = acc[i][j];
      int pc = p0 + ncol + j * 16 + n;
#pragma unroll
      for (int r = 0; r < 8; ++r) {
        int sr = s0 + srow + i * 16 + mh + r;
        Sx[(size_t)sr * 4096 + pc] = a.f[r];
      }
    }
}

// ---------------------------------------------------------------------------
// Kernel 3: masked scaled softmax over p for each query row s; emit bf16 yi[s][p].
// ---------------------------------------------------------------------------
__global__ void __launch_bounds__(256)
softmax_kernel(const float* __restrict__ Sx, const float* __restrict__ invn,
               const float* __restrict__ mmf, u16* __restrict__ yi)
{
  __shared__ float red[256];
  int s = blockIdx.x;
  int tid = threadIdx.x;
  const float* row = Sx + (size_t)s * 4096;

  float zmax = -3.4e38f;
  for (int p = tid; p < 4096; p += 256) {
    float z = row[p] * invn[p] * mmf[p] * 10.0f;
    zmax = fmaxf(zmax, z);
  }
  red[tid] = zmax; __syncthreads();
  for (int off = 128; off > 0; off >>= 1) {
    if (tid < off) red[tid] = fmaxf(red[tid], red[tid + off]);
    __syncthreads();
  }
  zmax = red[0]; __syncthreads();

  float zsum = 0.f;
  for (int p = tid; p < 4096; p += 256) {
    float z = row[p] * invn[p] * mmf[p] * 10.0f;
    zsum += __expf(z - zmax);
  }
  red[tid] = zsum; __syncthreads();
  for (int off = 128; off > 0; off >>= 1) {
    if (tid < off) red[tid] += red[tid + off];
    __syncthreads();
  }
  float rs = 1.0f / red[0];

  for (int p = tid; p < 4096; p += 256) {
    float z = row[p] * invn[p] * mmf[p] * 10.0f;
    float v = fmaxf(__expf(z - zmax) * rs * mmf[p], 1e-8f);
    yi[(size_t)s * 4096 + p] = f2bf(v);
  }
}

// ---------------------------------------------------------------------------
// Kernel 4: materialize x1 4x4/stride-2 patches as bf16 rw[tap(16)][c][p].
// ---------------------------------------------------------------------------
__global__ void __launch_bounds__(256)
rw_build_kernel(const float* __restrict__ x1, u16* __restrict__ rw)
{
  int c = blockIdx.x;          // 0..127
  int t = blockIdx.y;          // 0..15
  int ti = t >> 2, tj = t & 3;
  int tid = threadIdx.x;
  const float* xc = x1 + (size_t)c * 128 * 128;
  u16* dst = rw + ((size_t)t * 128 + c) * 4096;
  for (int p = tid; p < 4096; p += 256) {
    int py = p >> 6, px = p & 63;
    int r = 2 * py + ti - 1;     // pad=1
    int q = 2 * px + tj - 1;
    float v = 0.f;
    if ((unsigned)r < 128u && (unsigned)q < 128u) v = xc[(r << 7) + q];
    dst[p] = f2bf(v);
  }
}

// ---------------------------------------------------------------------------
// Kernel 5: stride-2 transposed conv as WMMA GEMM over p (K=4096, <=4 parity taps).
// ---------------------------------------------------------------------------
__global__ void __launch_bounds__(256)
deconv_kernel(const u16* __restrict__ yi, const u16* __restrict__ rw,
              float* __restrict__ yOut)   // [128][128][128] (one sample)
{
  int tid  = threadIdx.x;
  int lane = tid & 31;
  int wv   = tid >> 5;
  int id   = blockIdx.x;          // 1024 tiles
  int oh   = id >> 3;
  int rem  = id & 7;
  int rwp  = rem >> 2;            // ow parity
  int qw0  = (rem & 3) * 16;
  int qh = oh >> 1, rh = oh & 1;

  int djs[2], tis[2], dws[2], tjs[2];
  if (rh == 0) { djs[0] = -1; djs[1] = 0; tis[0] = 3; tis[1] = 1; }
  else         { djs[0] =  0; djs[1] = 1; tis[0] = 2; tis[1] = 0; }
  if (rwp == 0){ dws[0] = -1; dws[1] = 0; tjs[0] = 3; tjs[1] = 1; }
  else         { dws[0] =  0; dws[1] = 1; tjs[0] = 2; tjs[1] = 0; }

  int n   = lane & 15;
  int c0  = wv * 16;
  int alb = (lane < 16) ? 0 : 16;
  int bkb = (lane < 16) ? 0 : 16;

  v8f acc = {0.f,0.f,0.f,0.f,0.f,0.f,0.f,0.f};

  for (int p0 = 0; p0 < 4096; p0 += 32) {
#pragma unroll
    for (int a = 0; a < 2; ++a) {
      int ih = qh + djs[a];
      if ((unsigned)ih >= 64u) continue;           // block-uniform
#pragma unroll
      for (int b = 0; b < 2; ++b) {
        int iw = qw0 + n + dws[b];                 // per-lane
        Frag16 bf;
        if ((unsigned)iw < 64u) {
          const uint4* pb = (const uint4*)(yi + (size_t)(ih * 64 + iw) * 4096 + p0 + bkb);
          bf.q[0] = pb[0]; bf.q[1] = pb[1];
        } else {
          bf.q[0] = make_uint4(0u,0u,0u,0u);
          bf.q[1] = make_uint4(0u,0u,0u,0u);
        }
        int t = tis[a] * 4 + tjs[b];
        const u16* pa = rw + ((size_t)t * 128 + (c0 + n)) * 4096 + p0 + alb;
        Frag16 af;
        af.q[0] = *(const uint4*)pa;
        af.q[1] = *(const uint4*)(pa + 8);
        acc = __builtin_amdgcn_wmma_f32_16x16x32_bf16(
            false, af.v, false, bf.v, (short)0, acc, false, false);
      }
    }
  }

  int ow = 2 * (qw0 + n) + rwp;
  int mh = (lane < 16) ? 0 : 8;
  Acc8 a8; a8.v = acc;
#pragma unroll
  for (int r = 0; r < 8; ++r) {
    int c = c0 + mh + r;
    yOut[((size_t)c * 128 + oh) * 128 + ow] = a8.f[r] * 0.25f;
  }
}

// ---------------------------------------------------------------------------
// Kernel 6: four dilated 3x3 convs (r=1,2,4,8) + bias + ReLU (2% of FLOPs).
// ---------------------------------------------------------------------------
__global__ void __launch_bounds__(128)
final_conv_kernel(const float* __restrict__ ybuf,  // [2][128][128][128]
                  const float* __restrict__ fw0, const float* __restrict__ fb0,
                  const float* __restrict__ fw1, const float* __restrict__ fb1,
                  const float* __restrict__ fw2, const float* __restrict__ fb2,
                  const float* __restrict__ fw3, const float* __restrict__ fb3,
                  float* __restrict__ out)         // [2][64][128][128]
{
  int ow = threadIdx.x;
  int oh = blockIdx.x;
  int og = blockIdx.y;
  int b  = blockIdx.z;
  int g = og >> 4, oc = og & 15;
  const float* fw = (g == 0) ? fw0 : (g == 1) ? fw1 : (g == 2) ? fw2 : fw3;
  const float* fb = (g == 0) ? fb0 : (g == 1) ? fb1 : (g == 2) ? fb2 : fb3;
  int r = 1 << g;
  const float* yb = ybuf + (size_t)b * 128 * 128 * 128;
  float acc = fb[oc];
  for (int c = 0; c < 128; ++c) {
    const float* yc = yb + (size_t)c * 128 * 128;
    const float* wr = fw + ((size_t)oc * 128 + c) * 9;
#pragma unroll
    for (int kh = 0; kh < 3; ++kh) {
      int row = oh + r * (kh - 1);
      if ((unsigned)row >= 128u) continue;
#pragma unroll
      for (int kw = 0; kw < 3; ++kw) {
        int col = ow + r * (kw - 1);
        if ((unsigned)col >= 128u) continue;
        acc = fmaf(yc[(row << 7) + col], wr[kh * 3 + kw], acc);
      }
    }
  }
  out[(((size_t)b * 64 + og) * 128 + oh) * 128 + ow] = fmaxf(acc, 0.f);
}

// ---------------------------------------------------------------------------
extern "C" void kernel_launch(void* const* d_in, const int* in_sizes, int n_in,
                              void* d_out, int out_size, void* d_ws, size_t ws_size,
                              hipStream_t stream)
{
  (void)in_sizes; (void)n_in; (void)out_size; (void)ws_size;
  const float* x1   = (const float*)d_in[0];
  const float* x2   = (const float*)d_in[1];
  const float* mask = (const float*)d_in[2];
  const float* fw[4]; const float* fb[4];
  for (int g = 0; g < 4; ++g) {
    fw[g] = (const float*)d_in[3 + 2 * g];
    fb[g] = (const float*)d_in[4 + 2 * g];
  }

  char* ws = (char*)d_ws;
  float* ybuf = (float*)(ws);                          //  16,777,216 B  y (both samples)
  float* Sx   = (float*)(ws + 16777216);               //  67,108,864 B  scores (reused/sample)
  u16*   yi   = (u16*)  (ws + 83886080);               //  33,554,432 B  softmax (bf16)
  u16*   Bm   = (u16*)  (ws + 117440512);              //   9,437,184 B  im2col (bf16)
  u16*   rwb  = (u16*)  (ws + 126877696);              //  16,777,216 B  x1 patches (bf16)
  float* invn = (float*)(ws + 143654912);              //      16,384 B
  float* mmb  = (float*)(ws + 143671296);              //      16,384 B

  for (int b = 0; b < 2; ++b) {
    const float* x2b = x2   + (size_t)b * 128 * 64 * 64;
    const float* mb  = mask + (size_t)b * 64 * 64;
    const float* x1b = x1   + (size_t)b * 128 * 128 * 128;
    prep_kernel       <<<4096, 128, 0, stream>>>(x2b, mb, Bm, invn, mmb);
    gemm_scores_kernel<<<dim3(32, 32), 256, 0, stream>>>(Bm, Sx);
    softmax_kernel    <<<4096, 256, 0, stream>>>(Sx, invn, mmb, yi);
    rw_build_kernel   <<<dim3(128, 16), 256, 0, stream>>>(x1b, rwb);
    deconv_kernel     <<<1024, 256, 0, stream>>>(yi, rwb,
                         ybuf + (size_t)b * 128 * 128 * 128);
  }
  final_conv_kernel<<<dim3(128, 64, 2), 128, 0, stream>>>(
      ybuf, fw[0], fb[0], fw[1], fb[1], fw[2], fb[2], fw[3], fb[3], (float*)d_out);
}
